// Model_39676907887974
// MI455X (gfx1250) — matrix operand order repaired and checked by reading, not compiled, not run on hardware
//
#include <hip/hip_runtime.h>
#include <hip/hip_bf16.h>

// Fused 1x1-proj + image attention for MI455X (gfx1250, wave32, WMMA).
// One workgroup = one (b, o) pair x 128-row slab. k/v/q/p/a staged in LDS as
// bf16; both GEMMs run on v_wmma_f32_16x16x32_bf16 with f32 accumulation.
// q is pre-scaled by 1/W so the 'a' bias tile can initialize the WMMA C
// accumulator (no scalar-global epilogue); v and a are stored transposed so
// all fragment reads are contiguous ds_load_b128.

typedef __attribute__((ext_vector_type(16))) __bf16 v16bf;
typedef __attribute__((ext_vector_type(8)))  float  v8f;

#define BSZ    32
#define CIN    3
#define COUT   32
#define HH     256
#define WW     256
#define HWSZ   (HH * WW)
#define ROWS   128          // q rows handled per workgroup (8 waves x 16 rows)
#define PITCH  264          // bf16 LDS row pitch for 256-wide rows (+8 pad)
#define APITCH 136          // bf16 LDS row pitch for 128-wide rows (+8 pad)

__device__ __forceinline__ __bf16 f2bf(float f) {
    unsigned u = __builtin_bit_cast(unsigned, f);
    unsigned r = u + 0x7FFFu + ((u >> 16) & 1u);   // round-to-nearest-even
    unsigned short h = (unsigned short)(r >> 16);
    return __builtin_bit_cast(__bf16, h);
}
__device__ __forceinline__ float bf2f(__bf16 h) {
    unsigned u = (unsigned)__builtin_bit_cast(unsigned short, h) << 16;
    return __builtin_bit_cast(float, u);
}

__global__ __launch_bounds__(256, 1)
void fused_attn_f32bf16(const float* __restrict__ x1, const float* __restrict__ x2,
                        const float* __restrict__ Wq, const float* __restrict__ bq,
                        const float* __restrict__ Wk, const float* __restrict__ bk,
                        const float* __restrict__ Wv, const float* __restrict__ bv,
                        const float* __restrict__ Wa, const float* __restrict__ ba,
                        float* __restrict__ out)
{
    extern __shared__ __bf16 lds[];
    __bf16* kvb = lds;                              // [HH][PITCH]  k, later v^T
    __bf16* qpb = lds + HH * PITCH;                 // [ROWS][PITCH] q, later p
    __bf16* abt = lds + HH * PITCH + ROWS * PITCH;  // [WW][APITCH]  a^T[g][h]

    const int blk = blockIdx.x;
    const int hs  = (blk & 1) * ROWS;        // row-slab start
    const int o   = (blk >> 1) & (COUT - 1); // output channel
    const int b   = blk >> 6;                // batch

    const float wk0 = Wk[o*3+0], wk1 = Wk[o*3+1], wk2 = Wk[o*3+2], kb0 = bk[o];
    const float wv0 = Wv[o*3+0], wv1 = Wv[o*3+1], wv2 = Wv[o*3+2], vb0 = bv[o];
    const float wq0 = Wq[o*3+0], wq1 = Wq[o*3+1], wq2 = Wq[o*3+2], qb0 = bq[o];
    const float wa0 = Wa[o*3+0], wa1 = Wa[o*3+1], wa2 = Wa[o*3+2], ab0 = ba[o];

    const float* x1b = x1 + (size_t)b * CIN * HWSZ;
    const float* x2b = x2 + (size_t)b * CIN * HWSZ;

    const int tid = threadIdx.x;
    const float inv_scale = 1.0f / (float)WW;

    // ---- stage k projection (full image) into LDS as bf16 ----
    for (int i = tid; i < HWSZ; i += 256) {
        float c0 = x2b[i], c1 = x2b[HWSZ + i], c2 = x2b[2*HWSZ + i];
        kvb[(i >> 8) * PITCH + (i & 255)] = f2bf(wk0*c0 + wk1*c1 + wk2*c2 + kb0);
    }
    // ---- stage q (pre-scaled by 1/W) and a^T for this slab's 128 rows ----
    for (int i = tid; i < ROWS * WW; i += 256) {
        int idx = (hs << 8) + i;     // global pixel h*256 + w
        int r   = i >> 8;            // local row (h - hs)
        int cidx = i & 255;          // column (w, or g for the a-tile)
        float q0 = x1b[idx], q1 = x1b[HWSZ + idx], q2 = x1b[2*HWSZ + idx];
        qpb[r * PITCH + cidx] =
            f2bf((wq0*q0 + wq1*q1 + wq2*q2 + qb0) * inv_scale);
        float a0 = x2b[idx], a1 = x2b[HWSZ + idx], a2 = x2b[2*HWSZ + idx];
        abt[cidx * APITCH + r] = f2bf(wa0*a0 + wa1*a1 + wa2*a2 + ab0);
    }
    __syncthreads();

    const int wave  = tid >> 5;
    const int lane  = tid & 31;
    const int lhalf = lane >> 4;     // 0: lanes 0-15, 1: lanes 16-31
    const int l16   = lane & 15;
    const int rowb  = wave * 16;     // this wave's 16 rows within the slab

    // ---- load q A-fragments (resident for all 16 g-blocks) ----
    // A 16x32 bf16 k order used here (B matches): lane<16: M=lane, K=e (contiguous); lanes 16-31: +16
    v16bf qa[8];
    #pragma unroll
    for (int c = 0; c < 8; ++c) {
        #pragma unroll
        for (int e = 0; e < 16; ++e) {
            int K = c*32 + e + lhalf*16;
            qa[c][e] = qpb[(rowb + l16) * PITCH + K];
        }
    }

    // ---- GEMM 1: scores[h, g] = (q/W) . k^T + a  (C init = a-tile) ----
    v8f acc[16];
    #pragma unroll
    for (int gb = 0; gb < 16; ++gb) {
        // C-layout init from a^T: lane col g = gb*16+l16; rows j contiguous
        v8f s;
        #pragma unroll
        for (int j = 0; j < 8; ++j)
            s[j] = bf2f(abt[(gb*16 + l16) * APITCH + rowb + lhalf*8 + j]);
        #pragma unroll
        for (int c = 0; c < 8; ++c) {
            v16bf kf;   // B[K=w][N=g] = k[g][w]; N = l16, K = c*32 + e + lhalf*16
            #pragma unroll
            for (int e = 0; e < 16; ++e) {
                int K = c*32 + e + lhalf*16;
                kf[e] = kvb[(gb*16 + l16) * PITCH + K];
            }
            s = __builtin_amdgcn_wmma_f32_16x16x32_bf16(
                    false, qa[c], false, kf, (short)0, s, false, false);
        }
        acc[gb] = s;
    }

    // ---- softmax over g (rows live in VGPR j across 16-lane half-groups) ----
    float rmax[8], rsum[8], rinv[8];
    #pragma unroll
    for (int j = 0; j < 8; ++j) {
        float m = acc[0][j];
        #pragma unroll
        for (int gb = 1; gb < 16; ++gb) m = fmaxf(m, acc[gb][j]);
        m = fmaxf(m, __shfl_xor(m, 1, 32));
        m = fmaxf(m, __shfl_xor(m, 2, 32));
        m = fmaxf(m, __shfl_xor(m, 4, 32));
        m = fmaxf(m, __shfl_xor(m, 8, 32));
        rmax[j] = m;
        rsum[j] = 0.0f;
    }
    #pragma unroll
    for (int gb = 0; gb < 16; ++gb) {
        #pragma unroll
        for (int j = 0; j < 8; ++j) {
            float p = __expf(acc[gb][j] - rmax[j]);   // v_exp_f32
            acc[gb][j] = p;
            rsum[j] += p;
        }
    }
    #pragma unroll
    for (int j = 0; j < 8; ++j) {
        float s = rsum[j];
        s += __shfl_xor(s, 1, 32);
        s += __shfl_xor(s, 2, 32);
        s += __shfl_xor(s, 4, 32);
        s += __shfl_xor(s, 8, 32);
        rinv[j] = 1.0f / s;
    }

    // ---- write P tiles (bf16, unnormalized) into the q region ----
    #pragma unroll
    for (int gb = 0; gb < 16; ++gb) {
        #pragma unroll
        for (int j = 0; j < 8; ++j)
            qpb[(rowb + j + lhalf*8) * PITCH + gb*16 + l16] = f2bf(acc[gb][j]);
    }
    __syncthreads();   // all waves done reading k, all p written

    // ---- overwrite k region with TRANSPOSED v projection: v^T[w][g] ----
    for (int i = tid; i < HWSZ; i += 256) {
        float c0 = x2b[i], c1 = x2b[HWSZ + i], c2 = x2b[2*HWSZ + i];
        kvb[(i & 255) * PITCH + (i >> 8)] = f2bf(wv0*c0 + wv1*c1 + wv2*c2 + vb0);
    }
    __syncthreads();

    // ---- GEMM 2: out[h, w] = (p . v) * rinv ----
    v16bf pa[8];
    #pragma unroll
    for (int c = 0; c < 8; ++c) {
        #pragma unroll
        for (int e = 0; e < 16; ++e) {
            int K = c*32 + e + lhalf*16;   // K = g
            pa[c][e] = qpb[(rowb + l16) * PITCH + K];
        }
    }

    float* outp = out + (size_t)(b * COUT + o) * HWSZ;
    #pragma unroll
    for (int wb = 0; wb < 16; ++wb) {
        v8f oa = {};
        #pragma unroll
        for (int c = 0; c < 8; ++c) {
            v16bf vf;   // B[K=g][N=w] = v[g][w] = v^T[w][g]; contiguous in g
            #pragma unroll
            for (int e = 0; e < 16; ++e) {
                int K = c*32 + e + lhalf*16;
                vf[e] = kvb[(wb*16 + l16) * PITCH + K];
            }
            oa = __builtin_amdgcn_wmma_f32_16x16x32_bf16(
                     false, pa[c], false, vf, (short)0, oa, false, false);
        }
        #pragma unroll
        for (int j = 0; j < 8; ++j) {
            int h = hs + rowb + j + lhalf*8;
            outp[h*WW + wb*16 + l16] = oa[j] * rinv[j];
        }
    }
}

extern "C" void kernel_launch(void* const* d_in, const int* in_sizes, int n_in,
                              void* d_out, int out_size, void* d_ws, size_t ws_size,
                              hipStream_t stream) {
    const float* x1 = (const float*)d_in[0];
    const float* x2 = (const float*)d_in[1];
    const float* Wq = (const float*)d_in[2];
    const float* bq = (const float*)d_in[3];
    const float* Wk = (const float*)d_in[4];
    const float* bk = (const float*)d_in[5];
    const float* Wv = (const float*)d_in[6];
    const float* bv = (const float*)d_in[7];
    const float* Wa = (const float*)d_in[8];
    const float* ba = (const float*)d_in[9];
    float* outp = (float*)d_out;

    constexpr size_t shmem =
        ((size_t)HH * PITCH + (size_t)ROWS * PITCH + (size_t)WW * APITCH) *
        sizeof(__bf16);                                    // ~266 KB
    static_assert(shmem <= 320u * 1024u, "exceeds 320KB WGP LDS");
    (void)hipFuncSetAttribute(reinterpret_cast<const void*>(fused_attn_f32bf16),
                              hipFuncAttributeMaxDynamicSharedMemorySize,
                              (int)shmem);

    dim3 grid(BSZ * COUT * (HH / ROWS));   // 2048 workgroups
    dim3 block(256);                        // 8 wave32 waves
    hipLaunchKernelGGL(fused_attn_f32bf16, grid, block, shmem, stream,
                       x1, x2, Wq, bq, Wk, bk, Wv, bv, Wa, ba, outp);
}
